// FiringRouter_49752901157059
// MI455X (gfx1250) — matrix operand order repaired and checked by reading, not compiled, not run on hardware
//
#include <hip/hip_runtime.h>
#include <hip/hip_bf16.h>

// ---------------------------------------------------------------------------
// FiringRouter for MI455X (gfx1250), wave32 + WMMA bf16.
//   B=16384 tokens, H=1024, P=8 paths, K=2, THR=0.1
// Round 3: all GEMM inner loops double-buffered; gate kernel shares the A
// fragment across two n-tiles (2 WMMAs per LDS A load); paths kernel keeps
// M=32 tiles (2 WMMAs per global B load).
// ---------------------------------------------------------------------------

#define Btok 16384
#define Hdim 1024
#define Hhalf 512
#define Pn 8

typedef __attribute__((ext_vector_type(16))) __bf16 v16bf;
typedef __attribute__((ext_vector_type(8)))  __bf16 v8bf;
typedef __attribute__((ext_vector_type(8)))  float  v8f;

__device__ __forceinline__ v16bf cat16(v8bf lo, v8bf hi) {
  return __builtin_shufflevector(lo, hi, 0,1,2,3,4,5,6,7,8,9,10,11,12,13,14,15);
}

__device__ __forceinline__ v8f wmma_bf16(v16bf a, v16bf b, v8f c) {
  // D(16x16 f32) = A(16x32 bf16) * B(32x16 bf16) + C
  return __builtin_amdgcn_wmma_f32_16x16x32_bf16(
      /*neg_a=*/false, a, /*neg_b=*/false, b,
      /*c_mod=*/(short)0, c, /*reuse_a=*/false, /*reuse_b=*/false);
}

__device__ __forceinline__ float silu(float v) {
  return v / (1.0f + __expf(-v));
}

// ---------------------------------------------------------------------------
// Micro-kernel 1: M=32, one n-tile, K=1024.  A from LDS (32 x 1024 bf16),
// B from global (transposed weight row, contiguous K).  One B fragment feeds
// two WMMAs (two m-subtiles).  Double-buffered over k.
// ---------------------------------------------------------------------------
struct FragM32 {
  v8bf a0l, a0h, a1l, a1h;  // A fragments for m-subtile 0 and 1
  v16bf b;                  // shared B fragment
};

__device__ __forceinline__ void ld_frag_m32(FragM32& f, const __bf16* As,
                                            const __bf16* Bp, int ln, int hi, int kb) {
  f.a0l = *(const v8bf*)&As[ln * Hdim + kb + 16 * hi];
  f.a0h = *(const v8bf*)&As[ln * Hdim + kb + 8 + 16 * hi];
  f.a1l = *(const v8bf*)&As[(16 + ln) * Hdim + kb + 16 * hi];
  f.a1h = *(const v8bf*)&As[(16 + ln) * Hdim + kb + 8 + 16 * hi];
  f.b   = *(const v16bf*)(Bp + kb + 16 * hi);
}

__device__ __forceinline__ void do_wmma_m32(const FragM32& f, v8f& acc0, v8f& acc1) {
  acc0 = wmma_bf16(cat16(f.a0l, f.a0h), f.b, acc0);
  acc1 = wmma_bf16(cat16(f.a1l, f.a1h), f.b, acc1);
}

__device__ __forceinline__ void gemm_m32(const __bf16* As, const __bf16* Bp,
                                         int ln, int hi, v8f& acc0, v8f& acc1) {
  FragM32 f0, f1;
  ld_frag_m32(f0, As, Bp, ln, hi, 0);
  for (int kb = 0; kb < Hdim - 64; kb += 64) {
    ld_frag_m32(f1, As, Bp, ln, hi, kb + 32);   // prefetch while f0 consumed
    do_wmma_m32(f0, acc0, acc1);
    ld_frag_m32(f0, As, Bp, ln, hi, kb + 64);
    do_wmma_m32(f1, acc0, acc1);
  }
  ld_frag_m32(f1, As, Bp, ln, hi, Hdim - 32);
  do_wmma_m32(f0, acc0, acc1);
  do_wmma_m32(f1, acc0, acc1);
}

// ---------------------------------------------------------------------------
// Micro-kernel 2: M=16, two n-tiles sharing the A fragment, K=1024.
// A from LDS (16 x 1024 bf16), B0/B1 two transposed-weight rows.
// Double-buffered over k.
// ---------------------------------------------------------------------------
struct FragM16N2 {
  v8bf al, ah;    // shared A fragment
  v16bf b0, b1;   // B fragments for the two n-tiles
};

__device__ __forceinline__ void ld_frag_n2(FragM16N2& f, const __bf16* As,
                                           const __bf16* Bp0, const __bf16* Bp1,
                                           int ln, int hi, int kb) {
  f.al = *(const v8bf*)&As[ln * Hdim + kb + 16 * hi];
  f.ah = *(const v8bf*)&As[ln * Hdim + kb + 8 + 16 * hi];
  f.b0 = *(const v16bf*)(Bp0 + kb + 16 * hi);
  f.b1 = *(const v16bf*)(Bp1 + kb + 16 * hi);
}

__device__ __forceinline__ void do_wmma_n2(const FragM16N2& f, v8f& acc0, v8f& acc1) {
  const v16bf a = cat16(f.al, f.ah);
  acc0 = wmma_bf16(a, f.b0, acc0);
  acc1 = wmma_bf16(a, f.b1, acc1);
}

__device__ __forceinline__ void gemm_m16n2(const __bf16* As, const __bf16* Bp0,
                                           const __bf16* Bp1, int ln, int hi,
                                           v8f& acc0, v8f& acc1) {
  FragM16N2 f0, f1;
  ld_frag_n2(f0, As, Bp0, Bp1, ln, hi, 0);
  for (int kb = 0; kb < Hdim - 64; kb += 64) {
    ld_frag_n2(f1, As, Bp0, Bp1, ln, hi, kb + 32);
    do_wmma_n2(f0, acc0, acc1);
    ld_frag_n2(f0, As, Bp0, Bp1, ln, hi, kb + 64);
    do_wmma_n2(f1, acc0, acc1);
  }
  ld_frag_n2(f1, As, Bp0, Bp1, ln, hi, Hdim - 32);
  do_wmma_n2(f0, acc0, acc1);
  do_wmma_n2(f1, acc0, acc1);
}

// ---------------------------------------------------------------------------
// Kernel 0: f32 (R x C) row-major  ->  bf16 (C x R) row-major (transposed),
// batched over blockIdx.z.  Block = (32, 8).
// ---------------------------------------------------------------------------
__global__ __launch_bounds__(256)
void conv_transpose_kernel(const float* __restrict__ src, __bf16* __restrict__ dst,
                           int R, int C) {
  __shared__ float tile[32][33];
  const int b  = blockIdx.z;
  const float* s = src + (size_t)b * R * C;
  __bf16*      d = dst + (size_t)b * R * C;
  const int c0 = blockIdx.x * 32, r0 = blockIdx.y * 32;
  const int tx = threadIdx.x, ty = threadIdx.y;
  #pragma unroll
  for (int i = 0; i < 32; i += 8)
    tile[ty + i][tx] = s[(size_t)(r0 + ty + i) * C + (c0 + tx)];
  __syncthreads();
  #pragma unroll
  for (int i = 0; i < 32; i += 8)
    d[(size_t)(c0 + ty + i) * R + (r0 + tx)] = (__bf16)tile[tx][ty + i];
}

// ---------------------------------------------------------------------------
// Kernel 1: gate MLP + softmax + top-2 threshold routing.
// One block = 16 tokens, 256 threads (8 waves).  gw1t is bf16 (512 x 1024)
// transposed (i.e. [n][k] contiguous in k).  8 waves x 2 tile-pairs = 32
// n-tiles = 512 columns.
// ---------------------------------------------------------------------------
__global__ __launch_bounds__(256)
void gate_kernel(const float* __restrict__ x, const __bf16* __restrict__ gw1t,
                 const float* __restrict__ gb1, const float* __restrict__ gw2,
                 const float* __restrict__ gb2, float* __restrict__ gate_out,
                 float* __restrict__ w_out) {
  __shared__ __bf16 Xs[16][Hdim];    // 32 KB
  __shared__ __bf16 Hs[16][Hhalf];   // 16 KB
  __shared__ float  Lg[16][Pn];

  const int tid  = threadIdx.x;
  const int m0   = blockIdx.x * 16;
  const int wave = tid >> 5, lane = tid & 31;
  const int ln   = lane & 15, hi = lane >> 4;

  for (int i = tid; i < 16 * Hdim; i += 256) {
    const int m = i >> 10, k = i & (Hdim - 1);
    Xs[m][k] = (__bf16)x[(size_t)(m0 + m) * Hdim + k];
  }
  __syncthreads();

  // ---- GEMM1: H(16x512) = silu(X(16x1024) @ W1(1024x512) + b1), via WMMA ----
  #pragma unroll
  for (int t = 0; t < 2; ++t) {
    const int nb0 = (wave * 4 + 2 * t) * 16;         // first n-tile of the pair
    const int nb1 = nb0 + 16;                        // second n-tile
    v8f acc0 = {}, acc1 = {};
    gemm_m16n2(&Xs[0][0],
               gw1t + (size_t)(nb0 + ln) * Hdim,
               gw1t + (size_t)(nb1 + ln) * Hdim,
               ln, hi, acc0, acc1);
    const float bias0 = gb1[nb0 + ln];
    const float bias1 = gb1[nb1 + ln];
    #pragma unroll
    for (int r = 0; r < 8; ++r) {
      Hs[r + 8 * hi][nb0 + ln] = (__bf16)silu(acc0[r] + bias0);
      Hs[r + 8 * hi][nb1 + ln] = (__bf16)silu(acc1[r] + bias1);
    }
  }
  __syncthreads();

  // ---- GEMM2 (tiny, 16x8 from K=512): scalar dots ----
  if (tid < 128) {
    const int t = tid >> 3, j = tid & 7;
    float s = gb2[j];
    for (int k = 0; k < Hhalf; ++k)
      s += (float)Hs[t][k] * gw2[(size_t)k * Pn + j];
    Lg[t][j] = s;
  }
  __syncthreads();

  // ---- softmax + top-2 + firing threshold ----
  if (tid < 16) {
    const int t = tid;
    float mx = Lg[t][0];
    #pragma unroll
    for (int j = 1; j < Pn; ++j) mx = fmaxf(mx, Lg[t][j]);
    float e[Pn], sum = 0.f;
    #pragma unroll
    for (int j = 0; j < Pn; ++j) { e[j] = __expf(Lg[t][j] - mx); sum += e[j]; }
    float g[Pn];
    const float inv = 1.0f / sum;
    #pragma unroll
    for (int j = 0; j < Pn; ++j) g[j] = e[j] * inv;
    int i1 = 0;
    #pragma unroll
    for (int j = 1; j < Pn; ++j) if (g[j] > g[i1]) i1 = j;
    int i2 = (i1 == 0) ? 1 : 0;
    #pragma unroll
    for (int j = 0; j < Pn; ++j) if (j != i1 && g[j] > g[i2]) i2 = j;
    float w[Pn];
    #pragma unroll
    for (int j = 0; j < Pn; ++j) w[j] = 0.f;
    if (g[i1] > 0.1f) w[i1] = g[i1];
    if (g[i2] > 0.1f) w[i2] = g[i2];
    const size_t row = (size_t)(m0 + t) * Pn;
    #pragma unroll
    for (int j = 0; j < Pn; ++j) { gate_out[row + j] = g[j]; w_out[row + j] = w[j]; }
  }
}

// ---------------------------------------------------------------------------
// Kernel 2: per-path  Y = silu(X@Wp + bp);  Z = Y@Qp;  combined += w[b,p]*Z,
// then normalize + blend with x.
// One block = 32 tokens, 512 threads (16 waves x 4 n-tiles x 2 m-subtiles).
// Dynamic LDS: X tile (64 KB) + Y tile (64 KB) + weights, ~130 KB.
// ---------------------------------------------------------------------------
__global__ __launch_bounds__(512)
void paths_kernel(const float* __restrict__ x, const float* __restrict__ w_buf,
                  const __bf16* __restrict__ pwt, const float* __restrict__ pb,
                  const __bf16* __restrict__ qwt, const float* __restrict__ blend,
                  float* __restrict__ out) {
  extern __shared__ __align__(32) char smem[];
  __bf16* Xs    = (__bf16*)smem;                     // 32*1024 bf16
  __bf16* Ys    = Xs + 32 * Hdim;                    // 32*1024 bf16
  float*  Ws    = (float*)(Ys + 32 * Hdim);          // 32*8 f32
  int*    flags = (int*)(Ws + 32 * Pn);              // 8

  const int tid  = threadIdx.x;
  const int m0   = blockIdx.x * 32;
  const int wave = tid >> 5, lane = tid & 31;
  const int ln   = lane & 15, hi = lane >> 4;

  for (int i = tid; i < 32 * Hdim; i += 512) {
    const int m = i >> 10, k = i & (Hdim - 1);
    Xs[m * Hdim + k] = (__bf16)x[(size_t)(m0 + m) * Hdim + k];
  }
  if (tid < 32 * Pn) {
    Ws[tid] = w_buf[(size_t)m0 * Pn + tid];
  }
  __syncthreads();
  if (tid < Pn) {
    int f = 0;
    for (int t = 0; t < 32; ++t) f |= (Ws[t * Pn + tid] > 0.f) ? 1 : 0;
    flags[tid] = f;
  }
  __syncthreads();

  v8f cacc0[4] = {{}, {}, {}, {}};   // tokens 0..15 of tile
  v8f cacc1[4] = {{}, {}, {}, {}};   // tokens 16..31 of tile

  for (int p = 0; p < Pn; ++p) {
    if (!flags[p]) continue;    // block-uniform: all threads read same LDS flag

    const __bf16* Wp = pwt + (size_t)p * Hdim * Hdim;
    const __bf16* Qp = qwt + (size_t)p * Hdim * Hdim;

    // ---- pass 1: Y = silu(X @ Wp + bp) into LDS (bf16) ----
    #pragma unroll
    for (int t = 0; t < 4; ++t) {
      const int nb = (wave * 4 + t) * 16;
      v8f acc0 = {}, acc1 = {};
      gemm_m32(Xs, Wp + (size_t)(nb + ln) * Hdim, ln, hi, acc0, acc1);
      const float bias = pb[p * Hdim + nb + ln];
      #pragma unroll
      for (int r = 0; r < 8; ++r) {
        Ys[(r + 8 * hi) * Hdim + nb + ln]      = (__bf16)silu(acc0[r] + bias);
        Ys[(16 + r + 8 * hi) * Hdim + nb + ln] = (__bf16)silu(acc1[r] + bias);
      }
    }
    __syncthreads();

    // ---- pass 2: Z = Y @ Qp ; cacc += w[token,p] * Z ----
    v8f wv0, wv1;
    #pragma unroll
    for (int r = 0; r < 8; ++r) {
      wv0[r] = Ws[(r + 8 * hi) * Pn + p];
      wv1[r] = Ws[(16 + r + 8 * hi) * Pn + p];
    }

    #pragma unroll
    for (int t = 0; t < 4; ++t) {
      const int nb = (wave * 4 + t) * 16;
      v8f acc0 = {}, acc1 = {};
      gemm_m32(Ys, Qp + (size_t)(nb + ln) * Hdim, ln, hi, acc0, acc1);
      #pragma unroll
      for (int r = 0; r < 8; ++r) {
        cacc0[t][r] += wv0[r] * acc0[r];
        cacc1[t][r] += wv1[r] * acc1[r];
      }
    }
    __syncthreads();   // Ys reused by next path
  }

  // ---- normalize + blend ----
  const float alpha = 1.0f / (1.0f + __expf(-blend[0]));
  float tot0[8], tot1[8];
  #pragma unroll
  for (int r = 0; r < 8; ++r) {
    float s0 = 0.f, s1 = 0.f;
    #pragma unroll
    for (int j = 0; j < Pn; ++j) {
      s0 += Ws[(r + 8 * hi) * Pn + j];
      s1 += Ws[(16 + r + 8 * hi) * Pn + j];
    }
    tot0[r] = s0;
    tot1[r] = s1;
  }
  #pragma unroll
  for (int t = 0; t < 4; ++t) {
    const int nb = (wave * 4 + t) * 16;
    #pragma unroll
    for (int r = 0; r < 8; ++r) {
      {
        const int tok = r + 8 * hi;
        const size_t idx = (size_t)(m0 + tok) * Hdim + nb + ln;
        const float xv = x[idx];
        const float comb = (tot0[r] > 0.f) ? (cacc0[t][r] / tot0[r]) : xv;
        out[idx] = alpha * comb + (1.0f - alpha) * xv;
      }
      {
        const int tok = 16 + r + 8 * hi;
        const size_t idx = (size_t)(m0 + tok) * Hdim + nb + ln;
        const float xv = x[idx];
        const float comb = (tot1[r] > 0.f) ? (cacc1[t][r] / tot1[r]) : xv;
        out[idx] = alpha * comb + (1.0f - alpha) * xv;
      }
    }
  }
}

// ---------------------------------------------------------------------------
// Host-side launch
// ---------------------------------------------------------------------------
extern "C" void kernel_launch(void* const* d_in, const int* in_sizes, int n_in,
                              void* d_out, int out_size, void* d_ws, size_t ws_size,
                              hipStream_t stream) {
  const float* x     = (const float*)d_in[0];
  const float* gw1   = (const float*)d_in[1];
  const float* gb1   = (const float*)d_in[2];
  const float* gw2   = (const float*)d_in[3];
  const float* gb2   = (const float*)d_in[4];
  const float* pw    = (const float*)d_in[5];
  const float* pb    = (const float*)d_in[6];
  const float* qw    = (const float*)d_in[7];
  const float* blend = (const float*)d_in[8];

  // workspace layout (bytes):
  //   [0, 512K)          w_buf   f32 (B x P)
  //   [512K, 1.5M)       gw1t    bf16 (512 x 1024) transposed
  //   [1.5M, 17.5M)      pwt     bf16 (8 x 1024 x 1024) transposed per path
  //   [17.5M, 33.5M)     qwt     bf16 (8 x 1024 x 1024) transposed per path
  char* ws = (char*)d_ws;
  float*  w_buf = (float*)ws;
  __bf16* gw1t  = (__bf16*)(ws + (size_t)524288);
  __bf16* pwt   = (__bf16*)(ws + (size_t)1572864);
  __bf16* qwt   = (__bf16*)(ws + (size_t)18350080);

  const dim3 cb(32, 8, 1);
  conv_transpose_kernel<<<dim3(Hhalf / 32, Hdim / 32, 1),  cb, 0, stream>>>(gw1, gw1t, Hdim, Hhalf);
  conv_transpose_kernel<<<dim3(Hdim / 32,  Hdim / 32, Pn), cb, 0, stream>>>(pw,  pwt,  Hdim, Hdim);
  conv_transpose_kernel<<<dim3(Hdim / 32,  Hdim / 32, Pn), cb, 0, stream>>>(qw,  qwt,  Hdim, Hdim);

  float* out      = (float*)d_out;
  float* gate_out = out + (size_t)Btok * Hdim;

  gate_kernel<<<Btok / 16, 256, 0, stream>>>(x, gw1t, gb1, gw2, gb2, gate_out, w_buf);

  const size_t smem2 = (size_t)(32 * Hdim * 2) * 2   // Xs + Ys (bf16)
                     + (size_t)(32 * Pn * 4)         // Ws
                     + (size_t)(Pn * 4);             // flags
  paths_kernel<<<Btok / 32, 512, smem2, stream>>>(x, w_buf, pwt, pb, qwt, blend, out);
}
